// att_decoder_dependency_1194000908529
// MI455X (gfx1250) — compile-verified, matrix operand order checked by reading, not run on hardware
//
#include <hip/hip_runtime.h>
#include <stdint.h>

// Problem constants (reference: B=512, N=16, D=256, FEAT=63, E=120)
#define B_BATCH 512
#define N_NODES 16
#define D_DIM   256
#define FEAT    63
#define E_EDGES 120
#define G3      768          // 3*D gate width
#define TILE_M  16           // batch rows per workgroup
#define THREADS 256          // 8 waves (wave32)

typedef __attribute__((ext_vector_type(16))) _Float16 v16h;
typedef __attribute__((ext_vector_type(8)))  float    v8f;

union H16 { _Float16 h; unsigned short u; };

// ---------------------------------------------------------------------------
// Pack an fp32 [K x Nsrc] row-major weight matrix into f16 WMMA B fragments.
// Fragment (kt, nt) = 512 halfs, laid out [lane][elem 0..15]:
//   lane<16 : N = nt*16 + lane,     K = kt*32 + {0..7, 16..23}
//   lane>=16: N = nt*16 + lane-16,  K = kt*32 + {8..15, 24..31}
// elem e<8 -> K = base+e ; e>=8 -> K = 16+base+(e-8), base = (lane<16)?0:8.
// Columns beyond Nsrc are zero-padded (Npad multiple of 16).
// ---------------------------------------------------------------------------
__global__ void pack_w16(const float* __restrict__ src, unsigned short* __restrict__ dst,
                         int K, int Nsrc, int Npad) {
  const int ntile = Npad >> 4;
  const int total = (K >> 5) * ntile * 512;
  for (int idx = blockIdx.x * blockDim.x + threadIdx.x; idx < total;
       idx += gridDim.x * blockDim.x) {
    int frag   = idx >> 9;
    int within = idx & 511;
    int lane   = within >> 4;
    int e      = within & 15;
    int kt     = frag / ntile;
    int nt     = frag - kt * ntile;
    int base   = (lane < 16) ? 0 : 8;
    int kl     = (e < 8) ? (base + e) : (16 + base + (e - 8));
    int k      = kt * 32 + kl;
    int n      = nt * 16 + (lane & 15);
    float v    = (n < Nsrc) ? src[(size_t)k * Nsrc + n] : 0.0f;
    H16 cv; cv.h = (_Float16)v;
    dst[idx] = cv.u;
  }
}

// ---------------------------------------------------------------------------
// Batch-independent "compact" readout: 512 -> 256 -> 128 -> 256 (relu, relu, lin)
// ---------------------------------------------------------------------------
__global__ void content0_kernel(const float* __restrict__ zph, const float* __restrict__ iph,
                                const float* __restrict__ w0, const float* __restrict__ b0,
                                const float* __restrict__ w1, const float* __restrict__ b1,
                                const float* __restrict__ w2, const float* __restrict__ b2,
                                float* __restrict__ out256) {
  __shared__ float cat[512];
  __shared__ float h0[256];
  __shared__ float h1[128];
  const int t = threadIdx.x;
  cat[t]       = zph[t];
  cat[256 + t] = iph[t];
  __syncthreads();
  {
    float s = b0[t];
    for (int k = 0; k < 512; ++k) s += cat[k] * w0[(size_t)k * 256 + t];
    h0[t] = fmaxf(s, 0.0f);
  }
  __syncthreads();
  if (t < 128) {
    float s = b1[t];
    for (int k = 0; k < 256; ++k) s += h0[k] * w1[(size_t)k * 128 + t];
    h1[t] = fmaxf(s, 0.0f);
  }
  __syncthreads();
  {
    float s = b2[t];
    for (int k = 0; k < 128; ++k) s += h1[k] * w2[(size_t)k * 256 + t];
    out256[t] = s;
  }
}

// ---------------------------------------------------------------------------
// Per-wave GEMM: C[16 x Nw] = A[16 x K] * Bpacked[K x Nw]
//   A: fp32 in LDS, row-major stride K; converted to f16 A fragments on the fly
//   B: prepacked f16 fragments in global (L2-resident), 2x b128 per lane
//   C: fp32 in LDS, row-major stride Nw
// Wave `wave` handles n-tiles wave, wave+8, ...
// ---------------------------------------------------------------------------
__device__ __forceinline__ void wave_gemm(const float* __restrict__ A, int K,
                                          const unsigned short* __restrict__ Bpk,
                                          float* __restrict__ C, int Nw,
                                          int lane, int wave) {
  const int ntiles = Nw >> 4;
  const int ktiles = K >> 5;
  const int m      = lane & 15;
  const int kshift = (lane >> 4) << 3;     // 0 or 8
  const int rbase  = (lane >> 4) << 3;     // C rows 0..7 / 8..15
  const int ccol   = lane & 15;
  for (int nt = wave; nt < ntiles; nt += 8) {
    v8f acc = {0.f, 0.f, 0.f, 0.f, 0.f, 0.f, 0.f, 0.f};
    for (int kt = 0; kt < ktiles; ++kt) {
      // --- A fragment: 16-bit A 16x32 striping, two 8-elem chunks per lane ---
      const float* ar = A + (size_t)m * K + kt * 32 + kshift;
      v16h a;
#pragma unroll
      for (int e = 0; e < 8; ++e) {
        a[e]     = (_Float16)ar[e];
        a[e + 8] = (_Float16)ar[e + 16];
      }
      // --- B fragment: 32 bytes/lane, contiguous & 32B aligned ---
      const unsigned short* bp = Bpk + (((size_t)kt * ntiles + nt) << 9) + (lane << 4);
      union { uint4 q[2]; v16h h; } bf;
      bf.q[0] = *(const uint4*)(bp);
      bf.q[1] = *(const uint4*)(bp + 8);
      if (kt + 1 < ktiles)
        __builtin_prefetch(Bpk + (((size_t)(kt + 1) * ntiles + nt) << 9) + (lane << 4), 0, 0);
      acc = __builtin_amdgcn_wmma_f32_16x16x32_f16(false, a, false, bf.h,
                                                   (short)0, acc, false, false);
    }
#pragma unroll
    for (int r = 0; r < 8; ++r)
      C[(size_t)(r + rbase) * Nw + nt * 16 + ccol] = acc[r];
  }
}

// ---------------------------------------------------------------------------
// Edge readout (256->128->64->2) on h_upd, then softmax(2) + BCE accumulate.
// Reuses G1/G2 as scratch. eidx = position in the E=120 edge list.
// ---------------------------------------------------------------------------
__device__ __forceinline__ void edge_readout_bce(
    const float* __restrict__ h_upd, float* __restrict__ G1, float* __restrict__ G2,
    float* __restrict__ accb,
    const unsigned short* __restrict__ e0P, const unsigned short* __restrict__ e1P,
    const unsigned short* __restrict__ e2P,
    const float* __restrict__ eb0, const float* __restrict__ eb1, const float* __restrict__ eb2,
    const float* __restrict__ CON, int b0, int eidx, int tid, int lane, int wave) {
  wave_gemm(h_upd, 256, e0P, G1, 128, lane, wave);
  __syncthreads();
  for (int idx = tid; idx < 16 * 128; idx += THREADS) {
    int m = idx >> 7, c = idx & 127;
    G2[m * 128 + c] = fmaxf(G1[m * 128 + c] + eb0[c], 0.0f);
  }
  __syncthreads();
  wave_gemm(G2, 128, e1P, G1, 64, lane, wave);
  __syncthreads();
  for (int idx = tid; idx < 16 * 64; idx += THREADS) {
    int m = idx >> 6, c = idx & 63;
    G2[m * 64 + c] = fmaxf(G1[m * 64 + c] + eb1[c], 0.0f);
  }
  __syncthreads();
  wave_gemm(G2, 64, e2P, G1, 16, lane, wave);   // only 2 of 16 cols valid (padded)
  __syncthreads();
  if (tid < 32) {
    int m = tid >> 1, c = tid & 1;
    float l0 = G1[m * 16 + 0] + eb2[0];
    float l1 = G1[m * 16 + 1] + eb2[1];
    float mx = fmaxf(l0, l1);
    float x0 = __expf(l0 - mx), x1 = __expf(l1 - mx);
    float p  = ((c == 0) ? x0 : x1) / (x0 + x1);
    float lp  = fmaxf(__logf(p),   -100.0f);
    float l1m = fmaxf(log1pf(-p),  -100.0f);
    float cv  = CON[((size_t)(b0 + m) * E_EDGES + eidx) * 2 + c];
    accb[tid] -= cv * lp + (1.0f - cv) * l1m;
  }
  __syncthreads();
}

// ---------------------------------------------------------------------------
// Main persistent kernel: one workgroup per 16 batch rows, runs the entire
// 15-step outer / triangular-inner GRU chain with WMMA GEMMs.
// ---------------------------------------------------------------------------
__global__ void __launch_bounds__(THREADS)
att_main(const float* __restrict__ X, const float* __restrict__ CON,
         const unsigned short* __restrict__ decWiP, const unsigned short* __restrict__ decWhP,
         const unsigned short* __restrict__ updWiP, const unsigned short* __restrict__ updWhP,
         const unsigned short* __restrict__ e0P, const unsigned short* __restrict__ e1P,
         const unsigned short* __restrict__ e2P,
         const float* __restrict__ content0,
         const float* __restrict__ decBi, const float* __restrict__ decBh,
         const float* __restrict__ updBi, const float* __restrict__ updBh,
         const float* __restrict__ eb0, const float* __restrict__ eb1,
         const float* __restrict__ eb2,
         const float* __restrict__ inpW, const float* __restrict__ inpB,
         float* __restrict__ out) {
  extern __shared__ float smem[];
  float* content = smem;             // [16][256]  (also reused as MLP output)
  float* h_dec   = smem + 4096;      // [16][256]
  float* h_upd   = smem + 8192;      // [16][256]
  float* G1      = smem + 12288;     // [16][768]  gate / readout scratch
  float* G2      = smem + 24576;     // [16][768]  gate / readout scratch
  float* accb    = smem + 36864;     // [16][2]    BCE accumulators

  const int tid  = threadIdx.x;
  const int lane = tid & 31;
  const int wave = tid >> 5;
  const int b0   = blockIdx.x * TILE_M;

  for (int idx = tid; idx < 4096; idx += THREADS) {
    content[idx] = content0[idx & 255];   // identical rows initially
    h_dec[idx]   = 0.0f;
  }
  if (tid < 32) accb[tid] = 0.0f;
  __syncthreads();

  for (int i = 0; i < N_NODES - 1; ++i) {
    const int ebase = (i * (i + 1)) >> 1;

    // ---- dec GRU: h_dec = GRU(dec, content, h_dec) ----
    wave_gemm(content, D_DIM, decWiP, G1, G3, lane, wave);
    wave_gemm(h_dec,   D_DIM, decWhP, G2, G3, lane, wave);
    __syncthreads();
    for (int idx = tid; idx < 4096; idx += THREADS) {
      int m = idx >> 8, c = idx & 255;
      float az = G1[m*G3 + c]       + decBi[c]       + G2[m*G3 + c]       + decBh[c];
      float ar = G1[m*G3 + 256 + c] + decBi[256 + c] + G2[m*G3 + 256 + c] + decBh[256 + c];
      float z  = 1.0f / (1.0f + __expf(-az));
      float r  = 1.0f / (1.0f + __expf(-ar));
      float g  = tanhf(G1[m*G3 + 512 + c] + decBi[512 + c] +
                       r * (G2[m*G3 + 512 + c] + decBh[512 + c]));
      h_dec[idx] = (1.0f - z) * g + z * h_dec[idx];
    }
    __syncthreads();

    // ---- upd GRU with h=0: h_upd = GRU(upd, h_dec, 0) => gh = bh only ----
    wave_gemm(h_dec, D_DIM, updWiP, G1, G3, lane, wave);
    __syncthreads();
    for (int idx = tid; idx < 4096; idx += THREADS) {
      int m = idx >> 8, c = idx & 255;
      float az = G1[m*G3 + c]       + updBi[c]       + updBh[c];
      float ar = G1[m*G3 + 256 + c] + updBi[256 + c] + updBh[256 + c];
      float z  = 1.0f / (1.0f + __expf(-az));
      float r  = 1.0f / (1.0f + __expf(-ar));
      float g  = tanhf(G1[m*G3 + 512 + c] + updBi[512 + c] + r * updBh[512 + c]);
      h_upd[idx] = (1.0f - z) * g;   // h_old == 0
    }
    __syncthreads();

    // first row entry (m_step = 0) lands (after reversal) at ebase + i
    edge_readout_bce(h_upd, G1, G2, accb, e0P, e1P, e2P, eb0, eb1, eb2,
                     CON, b0, ebase + i, tid, lane, wave);

    for (int j = 0; j < i; ++j) {
      // ---- tiny MLP: xin = relu(x[:, i+1, 3+4j : 3+4j+2] @ inpW + inpB) ----
      for (int idx = tid; idx < 4096; idx += THREADS) {
        int m = idx >> 8, c = idx & 255;
        const float* xr = X + ((size_t)(b0 + m) * N_NODES + (i + 1)) * FEAT + 3 + 4 * j;
        content[idx] = fmaxf(xr[0] * inpW[c] + xr[1] * inpW[256 + c] + inpB[c], 0.0f);
      }
      __syncthreads();

      // ---- upd GRU: h_upd = GRU(upd, xin, h_upd) ----
      wave_gemm(content, D_DIM, updWiP, G1, G3, lane, wave);
      wave_gemm(h_upd,   D_DIM, updWhP, G2, G3, lane, wave);
      __syncthreads();
      for (int idx = tid; idx < 4096; idx += THREADS) {
        int m = idx >> 8, c = idx & 255;
        float az = G1[m*G3 + c]       + updBi[c]       + G2[m*G3 + c]       + updBh[c];
        float ar = G1[m*G3 + 256 + c] + updBi[256 + c] + G2[m*G3 + 256 + c] + updBh[256 + c];
        float z  = 1.0f / (1.0f + __expf(-az));
        float r  = 1.0f / (1.0f + __expf(-ar));
        float g  = tanhf(G1[m*G3 + 512 + c] + updBi[512 + c] +
                         r * (G2[m*G3 + 512 + c] + updBh[512 + c]));
        h_upd[idx] = (1.0f - z) * g + z * h_upd[idx];
      }
      __syncthreads();

      // row entry m_step = j+1 lands at ebase + i - (j+1)
      edge_readout_bce(h_upd, G1, G2, accb, e0P, e1P, e2P, eb0, eb1, eb2,
                       CON, b0, ebase + i - (j + 1), tid, lane, wave);
    }

    // content = h_upd for next outer iteration
    for (int idx = tid; idx < 4096; idx += THREADS) content[idx] = h_upd[idx];
    __syncthreads();
  }

  if (tid < 32) out[(size_t)(b0 + (tid >> 1)) * 2 + (tid & 1)] = accb[tid];
}

// ---------------------------------------------------------------------------
// Host launch. Input order follows setup_inputs() dict insertion order:
//  0:x 1:con 2:n 3:z_ph 4:init_ph
//  5..10: compact l0.w/l0.b/l1.w/l1.b/l2.w/l2.b
// 11..16: edge    l0.w/l0.b/l1.w/l1.b/l2.w/l2.b
// 17:inp.w 18:inp.b
// 19..22: dec wi/wh/bi/bh   23..26: upd wi/wh/bi/bh
// ---------------------------------------------------------------------------
extern "C" void kernel_launch(void* const* d_in, const int* in_sizes, int n_in,
                              void* d_out, int out_size, void* d_ws, size_t ws_size,
                              hipStream_t stream) {
  (void)in_sizes; (void)n_in; (void)out_size; (void)ws_size;

  const float* X     = (const float*)d_in[0];
  const float* CON   = (const float*)d_in[1];
  const float* zph   = (const float*)d_in[3];
  const float* iph   = (const float*)d_in[4];
  const float* cw0   = (const float*)d_in[5];
  const float* cb0   = (const float*)d_in[6];
  const float* cw1   = (const float*)d_in[7];
  const float* cb1   = (const float*)d_in[8];
  const float* cw2   = (const float*)d_in[9];
  const float* cb2   = (const float*)d_in[10];
  const float* ew0   = (const float*)d_in[11];
  const float* eb0   = (const float*)d_in[12];
  const float* ew1   = (const float*)d_in[13];
  const float* eb1   = (const float*)d_in[14];
  const float* ew2   = (const float*)d_in[15];
  const float* eb2   = (const float*)d_in[16];
  const float* inpW  = (const float*)d_in[17];
  const float* inpB  = (const float*)d_in[18];
  const float* decWi = (const float*)d_in[19];
  const float* decWh = (const float*)d_in[20];
  const float* decBi = (const float*)d_in[21];
  const float* decBh = (const float*)d_in[22];
  const float* updWi = (const float*)d_in[23];
  const float* updWh = (const float*)d_in[24];
  const float* updBi = (const float*)d_in[25];
  const float* updBh = (const float*)d_in[26];

  // Workspace layout: packed f16 weights, then content0 vector.
  unsigned short* pk = (unsigned short*)d_ws;
  size_t o = 0;
  unsigned short* decWiP = pk + o; o += (size_t)256 * 768;
  unsigned short* decWhP = pk + o; o += (size_t)256 * 768;
  unsigned short* updWiP = pk + o; o += (size_t)256 * 768;
  unsigned short* updWhP = pk + o; o += (size_t)256 * 768;
  unsigned short* e0Pp   = pk + o; o += (size_t)256 * 128;
  unsigned short* e1Pp   = pk + o; o += (size_t)128 * 64;
  unsigned short* e2Pp   = pk + o; o += (size_t)64  * 16;   // N padded 2 -> 16
  float* c0 = (float*)(pk + o);                             // 256 floats

  pack_w16<<<256, 256, 0, stream>>>(decWi, decWiP, 256, 768, 768);
  pack_w16<<<256, 256, 0, stream>>>(decWh, decWhP, 256, 768, 768);
  pack_w16<<<256, 256, 0, stream>>>(updWi, updWiP, 256, 768, 768);
  pack_w16<<<256, 256, 0, stream>>>(updWh, updWhP, 256, 768, 768);
  pack_w16<<< 64, 256, 0, stream>>>(ew0,   e0Pp,   256, 128, 128);
  pack_w16<<< 16, 256, 0, stream>>>(ew1,   e1Pp,   128,  64,  64);
  pack_w16<<<  4, 256, 0, stream>>>(ew2,   e2Pp,    64,   2,  16);

  content0_kernel<<<1, 256, 0, stream>>>(zph, iph, cw0, cb0, cw1, cb1, cw2, cb2, c0);

  const size_t lds_bytes = (size_t)36896 * sizeof(float);  // ~144 KB of 320 KB/WGP
  att_main<<<B_BATCH / TILE_M, THREADS, lds_bytes, stream>>>(
      X, CON, decWiP, decWhP, updWiP, updWhP, e0Pp, e1Pp, e2Pp, c0,
      decBi, decBh, updBi, updBh, eb0, eb1, eb2, inpW, inpB, (float*)d_out);
}
